// MultiHeadAttention_46076409151568
// MI455X (gfx1250) — matrix operand order repaired and checked by reading, not compiled, not run on hardware
//
#include <hip/hip_runtime.h>
#include <hip/hip_bf16.h>

// Problem constants (match the JAX reference)
#define BB   4
#define SS   2048
#define DD   1024
#define HH   16
#define DKK  64

typedef __attribute__((ext_vector_type(16))) __bf16 v16bf;
typedef __attribute__((ext_vector_type(8)))  float  v8f;

// ---------------------------------------------------------------------------
// fp32 -> bf16, round-to-nearest-even
// ---------------------------------------------------------------------------
__device__ __forceinline__ __bf16 f2bf(float f) {
  union { float f; unsigned u; } x; x.f = f;
  unsigned r = x.u + 0x7FFFu + ((x.u >> 16) & 1u);
  union { unsigned short s; __bf16 b; } y;
  y.s = (unsigned short)(r >> 16);
  return y.b;
}

__device__ __forceinline__ v8f wmma_bf16(v16bf a, v16bf b, v8f c) {
  // D = A(16x32 bf16) * B(32x16 bf16) + C(16x16 f32)
  return __builtin_amdgcn_wmma_f32_16x16x32_bf16(
      /*neg_a=*/false, a, /*neg_b=*/false, b,
      /*c_mod=*/(short)0, c, /*reuse_a=*/false, /*reuse_b=*/false);
}

__device__ __forceinline__ v8f v8f_zero() {
  v8f z = {0.f, 0.f, 0.f, 0.f, 0.f, 0.f, 0.f, 0.f};
  return z;
}

// ---------------------------------------------------------------------------
// CDNA5 async global->LDS copy (ASYNCcnt path, ISA 10.7 / 15.18.3).
// LDS address is taken from the VDST VGPR; generic shared pointers carry the LDS byte
// offset in their low 32 bits (aperture lives in addr[63:32]).
// ---------------------------------------------------------------------------
__device__ __forceinline__ void async_copy_b128(const __bf16* gsrc, __bf16* ldst) {
  unsigned lds_off = (unsigned)(size_t)ldst;
  asm volatile("global_load_async_to_lds_b128 %0, %1, off"
               :: "v"(lds_off), "v"(gsrc)
               : "memory");
}
__device__ __forceinline__ void wait_async_le4() {
  asm volatile("s_wait_asynccnt 0x4" ::: "memory");
}
__device__ __forceinline__ void wait_async_0() {
  asm volatile("s_wait_asynccnt 0x0" ::: "memory");
}

// ---------------------------------------------------------------------------
// Fragment loaders (ISA 7.12.2 layouts, wave32)
//
// A (16x32, 16-bit): lane m (m<16) holds row m, K = {kb+0..7, kb+16..23};
//                    lane m+16 holds row m, K = {kb+8..15, kb+24..31}.
// B (32x16, 16-bit): lane n (n<16) holds col n, K = {kb+0..7, kb+16..23};
//                    lane n+16 holds col n, K = {kb+8..15, kb+24..31}.
// C (16x16, f32)   : VGPR r, lanes 0-15 -> row r, col lane;
//                    lanes 16-31 -> row r+8, col lane-16.
// ---------------------------------------------------------------------------
__device__ __forceinline__ v16bf load_a_bf16(const __bf16* __restrict__ base,
                                             int ld, int m0, int kb, int lane) {
  const int row = m0 + (lane & 15);
  const int kh  = (lane >> 4) * 8;
  const __bf16* p = base + (size_t)row * ld + kb + kh;
  union { uint4 q[2]; v16bf v; } u;
  u.q[0] = *(const uint4*)(p);        // K = kb+kh .. +7
  u.q[1] = *(const uint4*)(p + 16);   // K = kb+16+kh .. +7
  return u.v;
}

__device__ __forceinline__ v16bf load_b_bf16(const __bf16* __restrict__ Bm,
                                             int ld, int n0, int kb, int lane) {
  const int n  = n0 + (lane & 15);
  const int ks = kb + 8 * (lane >> 4);
  const __bf16* p = Bm + (size_t)n * ld + ks;
  union { uint4 q[2]; v16bf v; } u;
  u.q[0] = *(const uint4*)(p);
  u.q[1] = *(const uint4*)(p + 16);
  return u.v;
}

// ---------------------------------------------------------------------------
// Kernel 0: fp32 -> bf16 bulk cast (n % 4 == 0)
// ---------------------------------------------------------------------------
__global__ __launch_bounds__(256) void cvt_f32_bf16_kernel(
    const float* __restrict__ src, __bf16* __restrict__ dst, int n4) {
  int i = blockIdx.x * blockDim.x + threadIdx.x;
  const int stride = gridDim.x * blockDim.x;
  for (; i < n4; i += stride) {
    float4 f = ((const float4*)src)[i];
    __bf16* d = dst + (size_t)i * 4;
    d[0] = f2bf(f.x); d[1] = f2bf(f.y); d[2] = f2bf(f.z); d[3] = f2bf(f.w);
  }
}

// ---------------------------------------------------------------------------
// Kernel 1: Y = X @ W^T + bias.  X:[M,1024] bf16 row-major, W:[1024,1024] bf16
// row-major (B-frag column n == W row n -> contiguous).
// Block 256 threads -> 128x128 output tile; K staged 32 at a time into LDS via
// double-buffered async global->LDS copies. Each wave: 16x128 (8 C-frags).
// PERM=true : store bf16 into [B,H,S,DK] head-major layout (QKV projections)
// PERM=false: store fp32 into [M,1024] (final output projection)
// ---------------------------------------------------------------------------
template <bool PERM>
__global__ __launch_bounds__(256) void gemm_bf16_kernel(
    const __bf16* __restrict__ A, const __bf16* __restrict__ Wb,
    const float* __restrict__ bias,
    __bf16* __restrict__ dst_perm, float* __restrict__ dst_f32) {
  __shared__ alignas(16) __bf16 Abuf[2][128 * 32];   // 2 x 8 KB
  __shared__ alignas(16) __bf16 Bbuf[2][128 * 32];   // 2 x 8 KB

  const int tid  = threadIdx.x;
  const int lane = tid & 31;
  const int wave = tid >> 5;
  const int n0 = blockIdx.x * 128;
  const int mB = blockIdx.y * 128;

  // Stage one 128x32 bf16 tile: 512 x b128 chunks, 2 per thread.
  auto issue_tile = [&](const __bf16* src, int r0, int k0, __bf16* dst) {
#pragma unroll
    for (int i = 0; i < 2; ++i) {
      const int c = tid + i * 256;          // 0..511
      const int row = c >> 2, c4 = c & 3;   // 4 x 16B per 32-elem row
      async_copy_b128(src + (size_t)(r0 + row) * DD + k0 + c4 * 8,
                      dst + row * 32 + c4 * 8);
    }
  };

  issue_tile(A,  mB, 0,  Abuf[0]);
  issue_tile(Wb, n0, 0,  Bbuf[0]);   // 4 outstanding
  issue_tile(A,  mB, 32, Abuf[1]);
  issue_tile(Wb, n0, 32, Bbuf[1]);   // 8 outstanding

  v8f acc[8];
#pragma unroll
  for (int t = 0; t < 8; ++t) acc[t] = v8f_zero();

  int cur = 0;
  for (int k0 = 0; k0 < DD; k0 += 32) {
    // Retire this buffer's async copies (front 4 of 8 in flight, in-order),
    // then barrier so every thread's copies are visible.
    if (k0 + 32 < DD) wait_async_le4(); else wait_async_0();
    __syncthreads();

    const v16bf a = load_a_bf16(Abuf[cur], 32, wave * 16, 0, lane);
#pragma unroll
    for (int nt = 0; nt < 8; ++nt) {
      v16bf b = load_b_bf16(Bbuf[cur], 32, nt * 16, 0, lane);
      acc[nt] = wmma_bf16(a, b, acc[nt]);
    }
    __syncthreads();   // all reads of buf[cur] done before it is refilled

    if (k0 + 64 < DD) {
      issue_tile(A,  mB, k0 + 64, Abuf[cur]);
      issue_tile(Wb, n0, k0 + 64, Bbuf[cur]);
    }
    cur ^= 1;
  }

  const int m0 = mB + wave * 16;
#pragma unroll
  for (int nt = 0; nt < 8; ++nt) {
    const int n  = n0 + nt * 16 + (lane & 15);
    const float bv = bias[n];
#pragma unroll
    for (int r = 0; r < 8; ++r) {
      const int m = m0 + r + 8 * (lane >> 4);
      const float v = acc[nt][r] + bv;
      if (PERM) {
        const int b = m >> 11, s = m & (SS - 1);       // m = b*S + s
        const int h = n >> 6, dk = n & (DKK - 1);      // n = h*DK + dk
        dst_perm[(((size_t)(b * HH + h)) * SS + s) * DKK + dk] = f2bf(v);
      } else {
        dst_f32[(size_t)m * DD + n] = v;
      }
    }
  }
}

// ---------------------------------------------------------------------------
// Kernel 2: flash attention over head-major Q/K/V ([B*H, S, DK] bf16).
// Block: 128 threads (4 waves) -> one (b,h) and 64 consecutive q rows.
// 64-wide K/V chunks: 16 WMMAs per softmax round (amortizes the per-chunk
// shuffle reductions, O-rescales and barriers over 2x the work vs 32-wide).
// K chunk staged via async global->LDS; V chunk transposed in-register.
// ---------------------------------------------------------------------------
__global__ __launch_bounds__(128) void flash_attn_kernel(
    const __bf16* __restrict__ Qh, const __bf16* __restrict__ Kh,
    const __bf16* __restrict__ Vh, __bf16* __restrict__ AO) {
  __shared__ alignas(16) __bf16 Kl[64 * 64];       // K chunk [sk][dk], 8 KB
  __shared__ alignas(16) __bf16 Vt[64 * 64];       // V chunk transposed [dk][sk], 8 KB
  __shared__ alignas(16) __bf16 Pb[4][16 * 64];    // per-wave P staging, 8 KB

  const int tid  = threadIdx.x;
  const int lane = tid & 31;
  const int wave = tid >> 5;
  const int bh = blockIdx.x;
  const int q0 = blockIdx.y * 64 + wave * 16;

  const __bf16* Qb = Qh + (size_t)bh * SS * DKK;
  const __bf16* Kb = Kh + (size_t)bh * SS * DKK;
  const __bf16* Vb = Vh + (size_t)bh * SS * DKK;

  // Q A-fragments for K-steps kb=0 and kb=32 (DK = 64), kept in registers.
  const v16bf qa0 = load_a_bf16(Qb, DKK, q0, 0, lane);
  const v16bf qa1 = load_a_bf16(Qb, DKK, q0, 32, lane);

  v8f o[4];
  float m[8], l[8];
#pragma unroll
  for (int t = 0; t < 4; ++t) o[t] = v8f_zero();
#pragma unroll
  for (int r = 0; r < 8; ++r) { m[r] = -1e30f; l[r] = 0.f; }

  const float scale = 0.125f;  // 1/sqrt(64)
  __bf16* P = &Pb[wave][0];

  for (int sk0 = 0; sk0 < SS; sk0 += 64) {
    // --- async-stage K chunk (64x64 bf16 = 512 x b128, 4 per thread) ------
#pragma unroll
    for (int i = 0; i < 4; ++i) {
      const int c = tid + i * 128;
      const int row = c >> 3, c8 = c & 7;
      async_copy_b128(Kb + (size_t)(sk0 + row) * DKK + c8 * 8,
                      &Kl[row * 64 + c8 * 8]);
    }
    // --- V chunk: load + in-register transpose into Vt[d][k] --------------
#pragma unroll
    for (int i = 0; i < 4; ++i) {
      const int c = tid + i * 128;
      const int row = c >> 3, c8 = c & 7;
      union { uint4 q; __bf16 e[8]; } vv;
      vv.q = *(const uint4*)(Vb + (size_t)(sk0 + row) * DKK + c8 * 8);
#pragma unroll
      for (int j = 0; j < 8; ++j) Vt[(c8 * 8 + j) * 64 + row] = vv.e[j];
    }
    wait_async_0();
    __syncthreads();

    // --- scores: S = Q (16x64) x K^T (64x64), four 16-col C tiles ---------
    v8f sc[4];
#pragma unroll
    for (int nt = 0; nt < 4; ++nt) {
      v8f c = v8f_zero();
      c = wmma_bf16(qa0, load_b_bf16(Kl, 64, nt * 16, 0,  lane), c);
      c = wmma_bf16(qa1, load_b_bf16(Kl, 64, nt * 16, 32, lane), c);
      sc[nt] = c;
    }

    // --- online softmax (row r lives in lanes {0..15} or {16..31});
    //     exp() results go straight into the per-wave P staging tile -------
#pragma unroll
    for (int r = 0; r < 8; ++r) {
      const float s0 = sc[0][r] * scale;
      const float s1 = sc[1][r] * scale;
      const float s2 = sc[2][r] * scale;
      const float s3 = sc[3][r] * scale;
      float mx = fmaxf(fmaxf(s0, s1), fmaxf(s2, s3));
      mx = fmaxf(mx, __shfl_xor(mx, 1));
      mx = fmaxf(mx, __shfl_xor(mx, 2));
      mx = fmaxf(mx, __shfl_xor(mx, 4));
      mx = fmaxf(mx, __shfl_xor(mx, 8));
      const float mn = fmaxf(m[r], mx);
      const float alpha = __expf(m[r] - mn);
      const float e0 = __expf(s0 - mn);
      const float e1 = __expf(s1 - mn);
      const float e2 = __expf(s2 - mn);
      const float e3 = __expf(s3 - mn);
      float rs = (e0 + e1) + (e2 + e3);
      rs += __shfl_xor(rs, 1);
      rs += __shfl_xor(rs, 2);
      rs += __shfl_xor(rs, 4);
      rs += __shfl_xor(rs, 8);
      l[r] = l[r] * alpha + rs;
      m[r] = mn;
      const int rowp = (r + 8 * (lane >> 4)) * 64 + (lane & 15);
      P[rowp]      = f2bf(e0);
      P[rowp + 16] = f2bf(e1);
      P[rowp + 32] = f2bf(e2);
      P[rowp + 48] = f2bf(e3);
#pragma unroll
      for (int t = 0; t < 4; ++t) o[t][r] = o[t][r] * alpha;
    }
    __syncthreads();

    // --- O += P (16x64) x V (64x64) ---------------------------------------
    const v16bf pa0 = load_a_bf16(P, 64, 0, 0,  lane);
    const v16bf pa1 = load_a_bf16(P, 64, 0, 32, lane);
#pragma unroll
    for (int t = 0; t < 4; ++t) {
      o[t] = wmma_bf16(pa0, load_b_bf16(Vt, 64, t * 16, 0,  lane), o[t]);
      o[t] = wmma_bf16(pa1, load_b_bf16(Vt, 64, t * 16, 32, lane), o[t]);
    }
    __syncthreads();  // before next chunk overwrites Kl/Vt
  }

  // --- normalize and store attn output as bf16 [B,S,D] --------------------
  const int b = bh >> 4, h = bh & 15;
#pragma unroll
  for (int t = 0; t < 4; ++t) {
    const int d = h * DKK + t * 16 + (lane & 15);
#pragma unroll
    for (int r = 0; r < 8; ++r) {
      const int s = q0 + r + 8 * (lane >> 4);
      const float inv = (l[r] > 0.f) ? (1.0f / l[r]) : 0.f;
      AO[((size_t)(b * SS + s)) * DD + d] = f2bf(o[t][r] * inv);
    }
  }
}

// ---------------------------------------------------------------------------
// Host-side launcher
// ---------------------------------------------------------------------------
extern "C" void kernel_launch(void* const* d_in, const int* in_sizes, int n_in,
                              void* d_out, int out_size, void* d_ws, size_t ws_size,
                              hipStream_t stream) {
  const float* query = (const float*)d_in[0];
  const float* key_  = (const float*)d_in[1];
  const float* value = (const float*)d_in[2];
  const float* Wq = (const float*)d_in[3]; const float* bq = (const float*)d_in[4];
  const float* Wk = (const float*)d_in[5]; const float* bk = (const float*)d_in[6];
  const float* Wv = (const float*)d_in[7]; const float* bv = (const float*)d_in[8];
  const float* Wo = (const float*)d_in[9]; const float* bo = (const float*)d_in[10];
  float* out = (float*)d_out;

  const size_t xel = (size_t)BB * SS * DD;   // 8,388,608
  const size_t wel = (size_t)DD * DD;        // 1,048,576

  // Workspace layout (bf16 elements), total ~120 MB:
  __bf16* p = (__bf16*)d_ws;
  __bf16* xq = p; p += xel;   // casted activations
  __bf16* xk = p; p += xel;
  __bf16* xv = p; p += xel;
  __bf16* wqb = p; p += wel;  // casted weights
  __bf16* wkb = p; p += wel;
  __bf16* wvb = p; p += wel;
  __bf16* wob = p; p += wel;
  __bf16* qh = p; p += xel;   // projected Q/K/V, head-major [B,H,S,DK]
  __bf16* kh = p; p += xel;
  __bf16* vh = p; p += xel;
  __bf16* ao = p; p += xel;   // attention output [B,S,D]

  // 1) fp32 -> bf16 casts
  cvt_f32_bf16_kernel<<<512, 256, 0, stream>>>(query, xq, (int)(xel / 4));
  cvt_f32_bf16_kernel<<<512, 256, 0, stream>>>(key_,  xk, (int)(xel / 4));
  cvt_f32_bf16_kernel<<<512, 256, 0, stream>>>(value, xv, (int)(xel / 4));
  cvt_f32_bf16_kernel<<<256, 256, 0, stream>>>(Wq, wqb, (int)(wel / 4));
  cvt_f32_bf16_kernel<<<256, 256, 0, stream>>>(Wk, wkb, (int)(wel / 4));
  cvt_f32_bf16_kernel<<<256, 256, 0, stream>>>(Wv, wvb, (int)(wel / 4));
  cvt_f32_bf16_kernel<<<256, 256, 0, stream>>>(Wo, wob, (int)(wel / 4));

  // 2) QKV projections (WMMA + double-buffered async LDS staging)
  const dim3 ggrid(DD / 128, (BB * SS) / 128);
  gemm_bf16_kernel<true><<<ggrid, 256, 0, stream>>>(xq, wqb, bq, qh, nullptr);
  gemm_bf16_kernel<true><<<ggrid, 256, 0, stream>>>(xk, wkb, bk, kh, nullptr);
  gemm_bf16_kernel<true><<<ggrid, 256, 0, stream>>>(xv, wvb, bv, vh, nullptr);

  // 3) Flash attention (WMMA scores + WMMA PV, online softmax, async K)
  flash_attn_kernel<<<dim3(BB * HH, SS / 64), 128, 0, stream>>>(qh, kh, vh, ao);

  // 4) Output projection (WMMA), fp32 store to d_out
  gemm_bf16_kernel<false><<<ggrid, 256, 0, stream>>>(ao, wob, bo, nullptr, out);
}
